// PredictiveModel_77068893160227
// MI455X (gfx1250) — matrix operand order repaired and checked by reading, not compiled, not run on hardware
//
#include <hip/hip_runtime.h>
#include <hip/hip_bf16.h>

// ---------------------------------------------------------------------------
// Model dims
#define B_   1024
#define L_   256
#define H_   64
#define M_   256
#define V_   64
#define HIST_ 4
#define LH_  32

// Recurrent kernel partitioning
#define NB_  32          // persistent blocks
#define RB_  32          // batch rows per block
#define NW_  8           // waves per block (256 threads, wave32)

typedef __attribute__((ext_vector_type(16))) __bf16 v16bf;
typedef __attribute__((ext_vector_type(8)))  __bf16 v8bf;
typedef __attribute__((ext_vector_type(8)))  float  v8f;
typedef __attribute__((ext_vector_type(4)))  float  v4f;

// ---------------------------------------------------------------------------
// WMMA operand builders (CDNA5 V_WMMA_F32_16X16X32_BF16 layouts, ISA 7.12.2)
//
// A (16x32, MxK), lane m=lane&15, half=lane>>4:
//   VGPR0..3 hold K = 8*half + {0..7}, VGPR4..7 hold K = 16 + 8*half + {0..7}
//   -> two contiguous 8-float chunks per lane; convert f32 -> bf16 inline.
__device__ __forceinline__ v16bf load_a_f32(const float* src, int lda,
                                            int m0, int k0, int lane) {
  const int half = (lane >> 4) & 1;
  const int m = lane & 15;
  const float* row = src + (m0 + m) * lda + k0;
  v4f x0 = *(const v4f*)(row + 8 * half);
  v4f x1 = *(const v4f*)(row + 8 * half + 4);
  v4f x2 = *(const v4f*)(row + 16 + 8 * half);
  v4f x3 = *(const v4f*)(row + 16 + 8 * half + 4);
  v16bf a;
#pragma unroll
  for (int i = 0; i < 4; i++) {
    a[i]      = (__bf16)x0[i];
    a[4 + i]  = (__bf16)x1[i];
    a[8 + i]  = (__bf16)x2[i];
    a[12 + i] = (__bf16)x3[i];
  }
  return a;
}

// B (32x16, KxN) from row-major weight W(OxK): B[k][n] = W[n][k].
// Same K order as A: elements 0..7 hold K = 8*half + {0..7}, 8..15 hold K = 16 + 8*half + {0..7}
// -> two 8-bf16 chunks per lane at W[n][k0 + 8*half ..] and W[n][k0 + 16 + 8*half ..]: two b128 loads.
__device__ __forceinline__ v16bf load_b_w(const __bf16* W, int ldw,
                                          int o0, int k0, int lane) {
  const int half = (lane >> 4) & 1;
  const int n = lane & 15;
  const __bf16* p = W + (o0 + n) * ldw + k0 + 8 * half;
  v8bf lo = *(const v8bf*)p;
  v8bf hi = *(const v8bf*)(p + 16);
  v16bf b;
#pragma unroll
  for (int i = 0; i < 8; i++) { b[i] = lo[i]; b[8 + i] = hi[i]; }
  return b;
}

// OUT[R x O] (f32, LDS) = A[R x K] (f32) @ W^T (W is OxK bf16) + bias
// R,O multiples of 16; K multiple of 32; (R/16)*(O/16) divisible by NW_
// so every wave runs the same trip count (EXEC all-ones for WMMA).
__device__ __forceinline__ void gemm_ld(const float* A, int lda,
                                        const __bf16* W, int ldw,
                                        const float* bias,
                                        float* OUT, int ldo,
                                        int R, int O, int K, int tid) {
  const int lane = tid & 31, wave = tid >> 5;
  const int ot_n = O >> 4;
  const int ntiles = (R >> 4) * ot_n;
  for (int tile = wave; tile < ntiles; tile += NW_) {
    const int rt = tile / ot_n, ot = tile - rt * ot_n;
    v8f acc = {};
    for (int k0 = 0; k0 < K; k0 += 32) {
      v16bf a = load_a_f32(A, lda, rt * 16, k0, lane);
      v16bf b = load_b_w(W, ldw, ot * 16, k0, lane);
      acc = __builtin_amdgcn_wmma_f32_16x16x32_bf16(false, a, false, b,
                                                    (short)0, acc, false, false);
    }
    const int n = lane & 15, half = (lane >> 4) & 1;
    const int col = ot * 16 + n;
    const float bv = bias ? bias[col] : 0.0f;
#pragma unroll
    for (int r = 0; r < 8; r++)
      OUT[(rt * 16 + r + 8 * half) * ldo + col] = acc[r] + bv;
  }
}

__device__ __forceinline__ float sigmoidf_(float x) {
  return 1.0f / (1.0f + __expf(-x));
}

// ---------------------------------------------------------------------------
// Kernel 0: zero the per-step sync slots (re-run every launch -> deterministic)
__global__ void init_sync(int* arrivals, float* probsum) {
  arrivals[threadIdx.x] = 0;
  probsum[threadIdx.x] = 0.0f;
}

// ---------------------------------------------------------------------------
// Kernel 1: pack weights f32 -> bf16 (row-major OxK), fuse LSTM ih|hh and bias
__global__ void pack_weights(const float* gwih_f, const float* gwhh_f,
                             const float* qw, const float* kw, const float* vw,
                             const float* lwih, const float* lwhh,
                             const float* lbih, const float* lbhh,
                             const float* headw_f,
                             __bf16* o_gwih, __bf16* o_gwhh,
                             __bf16* o_q, __bf16* o_k, __bf16* o_v,
                             __bf16* o_lcat, float* o_lbias, __bf16* o_head) {
  const int tid = blockIdx.x * blockDim.x + threadIdx.x;
  const int stride = gridDim.x * blockDim.x;
  for (int i = tid; i < 3 * H_ * 2 * H_; i += stride) o_gwih[i] = (__bf16)gwih_f[i];
  for (int i = tid; i < 3 * H_ * H_;     i += stride) o_gwhh[i] = (__bf16)gwhh_f[i];
  for (int i = tid; i < H_ * H_; i += stride) {
    o_q[i] = (__bf16)qw[i];
    o_k[i] = (__bf16)kw[i];
    o_v[i] = (__bf16)vw[i];
    o_head[i] = (__bf16)headw_f[i];
  }
  for (int i = tid; i < 4 * LH_ * (H_ + LH_); i += stride) {  // 128 x 96
    const int o = i / (H_ + LH_), k = i - o * (H_ + LH_);
    o_lcat[i] = (__bf16)(k < H_ ? lwih[o * H_ + k] : lwhh[o * LH_ + (k - H_)]);
  }
  for (int i = tid; i < 4 * LH_; i += stride) o_lbias[i] = lbih[i] + lbhh[i];
}

// ---------------------------------------------------------------------------
// Kernel 2: kmem/vmem = memory @ {k,v}_w.T + b    (262144 x 64 x 64, x2)
// One wave per 16-row strip; A operands built once, reused for K and V (8
// output tiles, 16 WMMA per wave). Output stored bf16 (halves attention BW,
// 64 MB total -> resident in 192 MB L2).
__global__ void __launch_bounds__(256) kv_proj(const float* mem,
                                               const __bf16* kwb, const __bf16* vwb,
                                               const float* kb, const float* vb,
                                               __bf16* kout, __bf16* vout) {
  const int lane = threadIdx.x & 31, wave = threadIdx.x >> 5;
  const long r0 = ((long)blockIdx.x * NW_ + wave) * 16;  // row base in B*M
  const float* abase = mem + r0 * H_;
  v16bf a0 = load_a_f32(abase, H_, 0, 0, lane);
  v16bf a1 = load_a_f32(abase, H_, 0, 32, lane);
  const int n = lane & 15, half = (lane >> 4) & 1;
#pragma unroll
  for (int sel = 0; sel < 2; sel++) {
    const __bf16* W = sel ? vwb : kwb;
    const float* bias = sel ? vb : kb;
    __bf16* out = sel ? vout : kout;
#pragma unroll
    for (int ot = 0; ot < 4; ot++) {
      v8f acc = {};
      acc = __builtin_amdgcn_wmma_f32_16x16x32_bf16(false, a0, false,
              load_b_w(W, H_, ot * 16, 0, lane), (short)0, acc, false, false);
      acc = __builtin_amdgcn_wmma_f32_16x16x32_bf16(false, a1, false,
              load_b_w(W, H_, ot * 16, 32, lane), (short)0, acc, false, false);
      const int col = ot * 16 + n;
      const float bv = bias[col];
#pragma unroll
      for (int r = 0; r < 8; r++)
        out[(r0 + r + 8 * half) * H_ + col] = (__bf16)(acc[r] + bv);
    }
  }
}

// ---------------------------------------------------------------------------
// Kernel 3: persistent recurrent kernel. 32 blocks x 256 threads, 32 batch
// rows each. One cross-grid sync per step (mean of trigger probs).
// LDS strides padded by +4 floats -> conflict-free b128 operand reads.
#define LDA_IN   132   // input  32 x [emb(64)|retr(64)] + pad
#define LDA_H     68
#define LDA_G    196   // GRU gates 3H + pad
#define LDA_XC   100   // LSTM cat input 96 + pad
#define LDA_LG   132   // LSTM gates 128 + pad
#define LDA_HL    36
#define LDA_SC   260   // scores M + pad

#define SM_INPUT  0
#define SM_H      (SM_INPUT + RB_ * LDA_IN)
#define SM_RETR   (SM_H + RB_ * LDA_H)
#define SM_HIST   (SM_RETR + RB_ * LDA_H)
#define SM_GI     (SM_HIST + HIST_ * RB_ * LDA_H)
#define SM_GH     (SM_GI + RB_ * LDA_G)
#define SM_XCAT   (SM_GH + RB_ * LDA_G)
#define SM_LG     (SM_XCAT + RB_ * LDA_XC)
#define SM_HL     (SM_LG + RB_ * LDA_LG)
#define SM_CL     (SM_HL + RB_ * LDA_HL)
#define SM_Q      (SM_CL + RB_ * LDA_HL)
#define SM_SC     (SM_Q + RB_ * LDA_H)
#define SM_RED    (SM_SC + RB_ * LDA_SC)
#define SM_RMAX   (SM_RED + RB_ * 8)
#define SM_RSUM   (SM_RMAX + RB_)
#define SM_BC     (SM_RSUM + RB_)
#define SM_TOTAL  (SM_BC + 4)

__global__ void __launch_bounds__(256, 1) recurrent(
    const int* seq, const float* embed_w,
    const float* gru_bih, const float* gru_bhh,
    const float* q_b, const float* trig_w, const float* trig_b,
    const float* head_b,
    const __bf16* gwih, const __bf16* gwhh, const __bf16* qwb,
    const __bf16* lcat, const float* lbias, const __bf16* headw,
    const __bf16* kmem, const __bf16* vmem,
    int* arrivals, float* probsum, float* out) {
  extern __shared__ float sm[];
  float* s_in   = sm + SM_INPUT;
  float* s_h    = sm + SM_H;
  float* s_rt   = sm + SM_RETR;
  float* s_hist = sm + SM_HIST;
  float* s_gi   = sm + SM_GI;
  float* s_gh   = sm + SM_GH;
  float* s_xc   = sm + SM_XCAT;
  float* s_lg   = sm + SM_LG;
  float* s_hl   = sm + SM_HL;
  float* s_cl   = sm + SM_CL;
  float* s_q    = sm + SM_Q;
  float* s_sc   = sm + SM_SC;
  float* s_red  = sm + SM_RED;
  float* s_rmax = sm + SM_RMAX;
  float* s_rsum = sm + SM_RSUM;
  float* s_bc   = sm + SM_BC;

  const int tid = threadIdx.x;
  const int rb0 = blockIdx.x * RB_;

  for (int i = tid; i < RB_ * LDA_H; i += 256) { s_h[i] = 0.0f; s_rt[i] = 0.0f; }
  __syncthreads();

  int reads = 0;
  for (int t = 0; t < L_; t++) {
    // ---- build GRU input = [embed(seq[b,t]) | retrieved] ----
    for (int i = tid; i < RB_ * H_; i += 256) {
      const int b = i >> 6, j = i & 63;
      const int tok = seq[(rb0 + b) * L_ + t];
      s_in[b * LDA_IN + j]      = embed_w[tok * H_ + j];
      s_in[b * LDA_IN + H_ + j] = s_rt[b * LDA_H + j];
    }
    __syncthreads();

    // ---- GRU gate GEMMs (WMMA bf16) ----
    gemm_ld(s_in, LDA_IN, gwih, 2 * H_, gru_bih, s_gi, LDA_G, RB_, 3 * H_, 2 * H_, tid);
    gemm_ld(s_h,  LDA_H,  gwhh, H_,     gru_bhh, s_gh, LDA_G, RB_, 3 * H_, H_,     tid);
    __syncthreads();

    // ---- GRU elementwise + history ring write ----
    const int slot = t & 3;
    for (int i = tid; i < RB_ * H_; i += 256) {
      const int b = i >> 6, j = i & 63;
      const float ir = s_gi[b * LDA_G + j],      hr = s_gh[b * LDA_G + j];
      const float iz = s_gi[b * LDA_G + 64 + j], hz = s_gh[b * LDA_G + 64 + j];
      const float in_ = s_gi[b * LDA_G + 128 + j], hn = s_gh[b * LDA_G + 128 + j];
      const float r = sigmoidf_(ir + hr);
      const float z = sigmoidf_(iz + hz);
      const float n = tanhf(in_ + r * hn);
      const float hv = (1.0f - z) * n + z * s_h[b * LDA_H + j];
      s_h[b * LDA_H + j] = hv;
      s_hist[slot * (RB_ * LDA_H) + b * LDA_H + j] = hv;
    }
    __syncthreads();

    if (t >= HIST_ - 1) {
      // ---- trigger LSTM over history (4 fused-K GEMMs) ----
      for (int i = tid; i < RB_ * LDA_HL; i += 256) { s_hl[i] = 0.0f; s_cl[i] = 0.0f; }
      __syncthreads();
      for (int js = 0; js < HIST_; js++) {
        const int sl = (t + 1 + js) & 3;
        for (int i = tid; i < RB_ * (H_ + LH_); i += 256) {
          const int b = i / (H_ + LH_), k = i - b * (H_ + LH_);
          s_xc[b * LDA_XC + k] = (k < H_) ? s_hist[sl * (RB_ * LDA_H) + b * LDA_H + k]
                                          : s_hl[b * LDA_HL + (k - H_)];
        }
        __syncthreads();
        gemm_ld(s_xc, LDA_XC, lcat, H_ + LH_, lbias, s_lg, LDA_LG,
                RB_, 4 * LH_, H_ + LH_, tid);
        __syncthreads();
        for (int i = tid; i < RB_ * LH_; i += 256) {
          const int b = i >> 5, j = i & 31;
          const float gi_ = s_lg[b * LDA_LG + j];
          const float gf  = s_lg[b * LDA_LG + 32 + j];
          const float gg  = s_lg[b * LDA_LG + 64 + j];
          const float go  = s_lg[b * LDA_LG + 96 + j];
          const float c = sigmoidf_(gf) * s_cl[b * LDA_HL + j]
                        + sigmoidf_(gi_) * tanhf(gg);
          s_cl[b * LDA_HL + j] = c;
          s_hl[b * LDA_HL + j] = sigmoidf_(go) * tanhf(c);
        }
        __syncthreads();
      }
      // ---- trigger prob + cross-grid mean (arrive counter, no full barrier) ----
      if (tid < RB_) {
        float acc = trig_b[0];
        for (int j = 0; j < LH_; j++) acc += s_hl[tid * LDA_HL + j] * trig_w[j];
        s_red[tid] = sigmoidf_(acc);
      }
      __syncthreads();
      if (tid == 0) {
        float ps = 0.0f;
        for (int b = 0; b < RB_; b++) ps += s_red[b];
        atomicAdd(&probsum[t], ps);
        __hip_atomic_fetch_add(&arrivals[t], 1, __ATOMIC_RELEASE,
                               __HIP_MEMORY_SCOPE_AGENT);
        while (__hip_atomic_load(&arrivals[t], __ATOMIC_ACQUIRE,
                                 __HIP_MEMORY_SCOPE_AGENT) < (int)gridDim.x)
          __builtin_amdgcn_s_sleep(2);
        s_bc[0] = __hip_atomic_load(&probsum[t], __ATOMIC_RELAXED,
                                    __HIP_MEMORY_SCOPE_AGENT);
      }
      __syncthreads();
      const int do_read = (s_bc[0] > 0.5f * (float)B_) ? 1 : 0;

      if (do_read) {
        reads++;
        // q = h @ q_w.T + q_b
        gemm_ld(s_h, LDA_H, qwb, H_, q_b, s_q, LDA_H, RB_, H_, H_, tid);
        __syncthreads();
        // scores = (q . kmem) * 1/sqrt(H)   (GEMV, L2-resident bf16)
        for (int i = tid; i < RB_ * M_; i += 256) {
          const int b = i >> 8, m = i & 255;
          const __bf16* kr = kmem + (((long)(rb0 + b)) * M_ + m) * H_;
          float acc = 0.0f;
          for (int j = 0; j < H_; j++) acc += s_q[b * LDA_H + j] * (float)kr[j];
          s_sc[b * LDA_SC + m] = acc * 0.125f;
        }
        __syncthreads();
        // row softmax: 8 threads per row
        {
          const int b = tid >> 3, sub = tid & 7;
          float mx = -1e30f;
          for (int m = sub; m < M_; m += 8) mx = fmaxf(mx, s_sc[b * LDA_SC + m]);
          s_red[b * 8 + sub] = mx;
        }
        __syncthreads();
        if (tid < RB_) {
          float mx = s_red[tid * 8];
          for (int k = 1; k < 8; k++) mx = fmaxf(mx, s_red[tid * 8 + k]);
          s_rmax[tid] = mx;
        }
        __syncthreads();
        {
          const int b = tid >> 3, sub = tid & 7;
          const float mx = s_rmax[b];
          float smm = 0.0f;
          for (int m = sub; m < M_; m += 8) {
            const float e = __expf(s_sc[b * LDA_SC + m] - mx);
            s_sc[b * LDA_SC + m] = e;
            smm += e;
          }
          s_red[b * 8 + sub] = smm;
        }
        __syncthreads();
        if (tid < RB_) {
          float smm = 0.0f;
          for (int k = 0; k < 8; k++) smm += s_red[tid * 8 + k];
          s_rsum[tid] = smm;
        }
        __syncthreads();
        // retrieved = (w . vmem) / rowsum
        for (int i = tid; i < RB_ * H_; i += 256) {
          const int b = i >> 6, hc = i & 63;
          const __bf16* vr = vmem + (((long)(rb0 + b)) * M_) * H_ + hc;
          float acc = 0.0f;
          for (int m = 0; m < M_; m++) acc += s_sc[b * LDA_SC + m] * (float)vr[m * H_];
          s_rt[b * LDA_H + hc] = acc / s_rsum[b];
        }
        __syncthreads();
      } else {
        for (int i = tid; i < RB_ * H_; i += 256) {
          const int b = i >> 6, j = i & 63;
          s_rt[b * LDA_H + j] = 0.0f;
        }
        __syncthreads();
      }
    }
  }

  // ---- output head: logits = h @ head_w.T + head_b ----
  gemm_ld(s_h, LDA_H, headw, H_, head_b, s_gi, LDA_G, RB_, V_, H_, tid);
  __syncthreads();
  for (int i = tid; i < RB_ * V_; i += 256) {
    const int b = i >> 6, v = i & 63;
    out[(rb0 + b) * V_ + v] = s_gi[b * LDA_G + v];
  }
  if (blockIdx.x == 0 && tid == 0)
    out[B_ * V_] = (float)reads / (float)L_;
}

// ---------------------------------------------------------------------------
// Workspace layout (bytes)
#define WS_ARRIVE   0
#define WS_PROBSUM  1024
#define WS_GWIH     4096                         // 192x128 bf16 = 49152
#define WS_GWHH     (WS_GWIH + 49152)            // 192x64  bf16 = 24576
#define WS_QW       (WS_GWHH + 24576)            // 64x64   bf16 = 8192
#define WS_KW       (WS_QW + 8192)
#define WS_VW       (WS_KW + 8192)
#define WS_LCAT     (WS_VW + 8192)               // 128x96 bf16 = 24576
#define WS_LBIAS    (WS_LCAT + 24576)            // 128 f32 = 512
#define WS_HEADW    (WS_LBIAS + 512)             // 8192
#define WS_KMEM     262144                       // B*M*H bf16 = 33554432
#define WS_VMEM     (WS_KMEM + 33554432)

extern "C" void kernel_launch(void* const* d_in, const int* in_sizes, int n_in,
                              void* d_out, int out_size, void* d_ws, size_t ws_size,
                              hipStream_t stream) {
  const int*   seq      = (const int*)d_in[0];
  const float* memory   = (const float*)d_in[1];
  const float* embed_w  = (const float*)d_in[2];
  const float* gru_wih  = (const float*)d_in[3];
  const float* gru_whh  = (const float*)d_in[4];
  const float* gru_bih  = (const float*)d_in[5];
  const float* gru_bhh  = (const float*)d_in[6];
  const float* q_w      = (const float*)d_in[7];
  const float* q_b      = (const float*)d_in[8];
  const float* k_w      = (const float*)d_in[9];
  const float* k_b      = (const float*)d_in[10];
  const float* v_w      = (const float*)d_in[11];
  const float* v_b      = (const float*)d_in[12];
  const float* lstm_wih = (const float*)d_in[13];
  const float* lstm_whh = (const float*)d_in[14];
  const float* lstm_bih = (const float*)d_in[15];
  const float* lstm_bhh = (const float*)d_in[16];
  const float* trig_w   = (const float*)d_in[17];
  const float* trig_b   = (const float*)d_in[18];
  const float* head_w   = (const float*)d_in[19];
  const float* head_b   = (const float*)d_in[20];

  char* ws = (char*)d_ws;
  int*    arrivals = (int*)(ws + WS_ARRIVE);
  float*  probsum  = (float*)(ws + WS_PROBSUM);
  __bf16* gwih  = (__bf16*)(ws + WS_GWIH);
  __bf16* gwhh  = (__bf16*)(ws + WS_GWHH);
  __bf16* qwb   = (__bf16*)(ws + WS_QW);
  __bf16* kwb   = (__bf16*)(ws + WS_KW);
  __bf16* vwb   = (__bf16*)(ws + WS_VW);
  __bf16* lcat  = (__bf16*)(ws + WS_LCAT);
  float*  lbias = (float*)(ws + WS_LBIAS);
  __bf16* headw = (__bf16*)(ws + WS_HEADW);
  __bf16* kmem  = (__bf16*)(ws + WS_KMEM);
  __bf16* vmem  = (__bf16*)(ws + WS_VMEM);

  const size_t smem_bytes = (size_t)SM_TOTAL * sizeof(float);
  hipFuncSetAttribute((const void*)recurrent,
                      hipFuncAttributeMaxDynamicSharedMemorySize,
                      (int)smem_bytes);

  init_sync<<<1, 256, 0, stream>>>(arrivals, probsum);

  pack_weights<<<64, 256, 0, stream>>>(gru_wih, gru_whh, q_w, k_w, v_w,
                                       lstm_wih, lstm_whh, lstm_bih, lstm_bhh,
                                       head_w,
                                       gwih, gwhh, qwb, kwb, vwb,
                                       lcat, lbias, headw);

  kv_proj<<<(B_ * M_ / 16) / NW_, 256, 0, stream>>>(memory, kwb, vwb, k_b, v_b,
                                                    kmem, vmem);

  recurrent<<<NB_, 256, smem_bytes, stream>>>(
      seq, embed_w, gru_bih, gru_bhh, q_b, trig_w, trig_b, head_b,
      gwih, gwhh, qwb, lcat, lbias, headw, kmem, vmem,
      arrivals, probsum, (float*)d_out);
}
